// PPM_50173807952180
// MI455X (gfx1250) — matrix operand order repaired and checked by reading, not compiled, not run on hardware
//
#include <hip/hip_runtime.h>
#include <math.h>

// ---------------------------------------------------------------------------
// PPM attention, fused, bf16 WMMA (gfx1250 / CDNA5, wave32)
// N=16, C=256, H=W=56, HW=3136
// ---------------------------------------------------------------------------

typedef __bf16 bf16;
typedef __attribute__((ext_vector_type(16))) __bf16 v16bf;
typedef __attribute__((ext_vector_type(8)))  __bf16 v8bf;
typedef __attribute__((ext_vector_type(8)))  float  v8f;

static constexpr int kN  = 16;
static constexpr int kC  = 256;
static constexpr int kHW = 3136;

// ---- WMMA fragment loaders (layouts per CDNA5 ISA 7.12.2) ------------------
// A-matrix 16x32 bf16, row-major source: lane(m=L%16, half=L/16) holds
//   K = half*8 + [0..7]  and  K = 16 + half*8 + [0..7]
__device__ __forceinline__ v16bf frag_a(const bf16* row_kbase, int half) {
  v8bf lo = *(const v8bf*)(row_kbase + half * 8);
  v8bf hi = *(const v8bf*)(row_kbase + 16 + half * 8);
  return __builtin_shufflevector(lo, hi, 0,1,2,3,4,5,6,7,8,9,10,11,12,13,14,15);
}
// B-matrix 32x16 bf16, loaded from B^T rows (n-major, K contiguous):
// lane(n=L%16) holds K = half*8 + [0..7]  and  K = 16 + half*8 + [0..7]
__device__ __forceinline__ v16bf frag_b(const bf16* bt_row_kbase, int half) {
  v8bf lo = *(const v8bf*)(bt_row_kbase + half * 8);
  v8bf hi = *(const v8bf*)(bt_row_kbase + 16 + half * 8);
  return __builtin_shufflevector(lo, hi, 0,1,2,3,4,5,6,7,8,9,10,11,12,13,14,15);
}

__device__ __forceinline__ v8f wmma_bf16(v16bf a, v16bf b, v8f c) {
  return __builtin_amdgcn_wmma_f32_16x16x32_bf16(false, a, false, b,
                                                 (short)0, c, false, false);
}

// ---------------------------------------------------------------------------
// K0: convert W (f32 [C,C]) -> bf16 (L2-resident operand)
// ---------------------------------------------------------------------------
__global__ void k_cvt_w(const float* __restrict__ Wf, bf16* __restrict__ Wb) {
  int i = blockIdx.x * 256 + threadIdx.x;
  Wb[i] = (bf16)Wf[i];
}

// ---------------------------------------------------------------------------
// K1: per-pixel channel L2 norm of feat; emit
//   fnT[n][p][c] = normalized feat (transposed)  -> A and B^T of score GEMM
//   ftT[n][p][c] = raw feat bf16 (transposed)    -> B^T of conv GEMM
// ---------------------------------------------------------------------------
__global__ void k_prep(const float* __restrict__ feat,
                       bf16* __restrict__ fnT, bf16* __restrict__ ftT) {
  int n = blockIdx.y;
  int p = blockIdx.x * 256 + threadIdx.x;
  if (p >= kHW) return;
  const float* f = feat + (size_t)n * kC * kHW + p;
  float ss = 0.f;
  for (int c = 0; c < kC; ++c) { float v = f[(size_t)c * kHW]; ss += v * v; }
  float inv = 1.f / fmaxf(sqrtf(ss), 1e-12f);
  bf16* on = fnT + ((size_t)n * kHW + p) * kC;
  bf16* ot = ftT + ((size_t)n * kHW + p) * kC;
  for (int c = 0; c < kC; ++c) {
    float v = f[(size_t)c * kHW];
    on[c] = (bf16)(v * inv);
    ot[c] = (bf16)v;
  }
}

// ---------------------------------------------------------------------------
// K2: 1x1 conv as WMMA GEMM fv[o,p] = sum_c W[o,c]*feat[c,p] + b[o],
// then channel-L2-normalize per p (block-wide LDS reduction), store
// fvn[n][o][hw] bf16 (row-major in hw => A-operand layout for output GEMM).
// Block: 256 threads (8 waves); p-tile = 16; wave w owns o = w*32..w*32+31.
// ---------------------------------------------------------------------------
__global__ void k_conv(const bf16* __restrict__ Wb, const bf16* __restrict__ ftT,
                       const float* __restrict__ bias, bf16* __restrict__ fvn) {
  int n    = blockIdx.y;
  int pb   = blockIdx.x * 16;
  int lane = threadIdx.x & 31;
  int wv   = threadIdx.x >> 5;
  int lm   = lane & 15, half = lane >> 4;
  int ob   = wv * 32;
  const bf16* ft_n = ftT + (size_t)n * kHW * kC;

  __shared__ float nrm2[16];
  if (threadIdx.x < 16) nrm2[threadIdx.x] = 0.f;
  __syncthreads();

  v8f acc[2] = {};
#pragma unroll
  for (int kk = 0; kk < 8; ++kk) {
    v16bf bfr = frag_b(ft_n + (size_t)(pb + lm) * kC + kk * 32, half);
#pragma unroll
    for (int os = 0; os < 2; ++os) {
      v16bf af = frag_a(Wb + (size_t)(ob + os * 16 + lm) * kC + kk * 32, half);
      acc[os] = wmma_bf16(af, bfr, acc[os]);
    }
  }
  // bias + per-lane partial sum of squares (D layout: VGPR i -> o = i + 8*half)
  float ss = 0.f;
#pragma unroll
  for (int os = 0; os < 2; ++os)
#pragma unroll
    for (int i = 0; i < 8; ++i) {
      int o = ob + os * 16 + i + 8 * half;
      float v = acc[os][i] + bias[o];
      acc[os][i] = v;
      ss += v * v;
    }
  atomicAdd(&nrm2[lm], ss);
  __syncthreads();
  float inv = 1.f / fmaxf(sqrtf(nrm2[lm]), 1e-12f);

  bf16* out_n = fvn + (size_t)n * kC * kHW;
#pragma unroll
  for (int os = 0; os < 2; ++os)
#pragma unroll
    for (int i = 0; i < 8; ++i) {
      int o = ob + os * 16 + i + 8 * half;
      out_n[(size_t)o * kHW + pb + lm] = (bf16)(acc[os][i] * inv);
    }
}

// ---------------------------------------------------------------------------
// K3: fused attention. Block (8 waves) owns p-tile of 32 and ALL 256 c.
// Sweep q in steps of 64:
//  phase1: wave (psub=w>>2, qsub=w&3) computes S16x16 = Fp.Fq (K=256, 8 WMMA),
//          relu^2 -> bf16 -> LDS S[32][64]
//  phase2: wave w accumulates O[c=w*32..+31][p 0..31] += fvn * S^T
//          (A from global fvn rows, B^T = LDS S rows, 8 WMMA)
// ---------------------------------------------------------------------------
__global__ void __launch_bounds__(256)
k_ppm(const bf16* __restrict__ fnT, const bf16* __restrict__ fvn,
      float* __restrict__ out) {
  int n     = blockIdx.y;
  int pbase = blockIdx.x * 32;
  int lane  = threadIdx.x & 31;
  int wv    = threadIdx.x >> 5;
  int lm    = lane & 15, half = lane >> 4;
  int psub1 = wv >> 2;      // phase-1 p sub-tile (0..1)
  int qsub  = wv & 3;       // phase-1 q sub-tile (0..3)
  int cb    = wv * 32;      // phase-2 c rows

  const bf16* fn_n = fnT + (size_t)n * kHW * kC;
  const bf16* fv_n = fvn + (size_t)n * kC * kHW;

  __shared__ __align__(16) bf16 smS[32 * 64];

  // Preload A1: this wave's 16 p-rows, full K=256 (8 fragments, 64 VGPRs)
  v16bf a1[8];
  {
    const bf16* arow = fn_n + (size_t)(pbase + psub1 * 16 + lm) * kC;
#pragma unroll
    for (int kk = 0; kk < 8; ++kk) a1[kk] = frag_a(arow + kk * 32, half);
  }

  v8f O[2][2] = {};  // [csub][psub]

  for (int qb = 0; qb < kHW; qb += 64) {
    // ---- phase 1: score tile ----
    v8f s = {};
    const bf16* brow = fn_n + (size_t)(qb + qsub * 16 + lm) * kC;
#pragma unroll
    for (int kk = 0; kk < 8; ++kk) {
      v16bf bfr = frag_b(brow + kk * 32, half);
      s = wmma_bf16(a1[kk], bfr, s);
    }
#pragma unroll
    for (int i = 0; i < 8; ++i) {
      float t = fmaxf(s[i], 0.f);
      t *= t;                                    // relu(att)^SHARPNESS, SHARPNESS=2
      int pr = psub1 * 16 + i + 8 * half;        // D row
      smS[pr * 64 + qsub * 16 + lm] = (bf16)t;   // D col = lane
    }
    __syncthreads();

    // ---- phase 2: output accumulation ----
#pragma unroll
    for (int ks = 0; ks < 2; ++ks) {
      int qk = ks * 32;
      v16bf a2[2];
#pragma unroll
      for (int cs = 0; cs < 2; ++cs)
        a2[cs] = frag_a(fv_n + (size_t)(cb + cs * 16 + lm) * kHW + qb + qk, half);
#pragma unroll
      for (int ps = 0; ps < 2; ++ps) {
        v16bf bs = frag_b(&smS[(ps * 16 + lm) * 64 + qk], half);
#pragma unroll
        for (int cs = 0; cs < 2; ++cs)
          O[cs][ps] = wmma_bf16(a2[cs], bs, O[cs][ps]);
      }
    }
    __syncthreads();  // protect LDS tile before next phase-1 writes
  }

  // ---- store (coalesced across lanes: p contiguous) ----
  float* out_n = out + (size_t)n * kC * kHW;
#pragma unroll
  for (int cs = 0; cs < 2; ++cs)
#pragma unroll
    for (int ps = 0; ps < 2; ++ps)
#pragma unroll
      for (int i = 0; i < 8; ++i) {
        int c = cb + cs * 16 + i + 8 * half;
        int p = pbase + ps * 16 + lm;
        out_n[(size_t)c * kHW + p] = O[cs][ps][i];
      }
}

// ---------------------------------------------------------------------------
extern "C" void kernel_launch(void* const* d_in, const int* in_sizes, int n_in,
                              void* d_out, int out_size, void* d_ws, size_t ws_size,
                              hipStream_t stream) {
  const float* feat = (const float*)d_in[0];   // [16,256,56,56] f32
  const float* Wf   = (const float*)d_in[1];   // [256,256] f32
  const float* bias = (const float*)d_in[2];   // [256] f32
  float* out        = (float*)d_out;           // [16,256,56,56] f32

  char* ws = (char*)d_ws;
  size_t szW = (size_t)kC * kC * sizeof(bf16);                 // 128 KB
  size_t szT = (size_t)kN * kHW * kC * sizeof(bf16);           // ~25.7 MB each
  size_t off = 0;
  bf16* Wb  = (bf16*)(ws + off); off += (szW + 511) & ~(size_t)511;
  bf16* fnT = (bf16*)(ws + off); off += (szT + 511) & ~(size_t)511;
  bf16* ftT = (bf16*)(ws + off); off += (szT + 511) & ~(size_t)511;
  bf16* fvn = (bf16*)(ws + off); off += (szT + 511) & ~(size_t)511;
  (void)ws_size; (void)in_sizes; (void)n_in; (void)out_size;

  k_cvt_w<<<dim3((kC * kC) / 256), 256, 0, stream>>>(Wf, Wb);
  k_prep <<<dim3((kHW + 255) / 256, kN), 256, 0, stream>>>(feat, fnT, ftT);
  k_conv <<<dim3(kHW / 16, kN), 256, 0, stream>>>(Wb, ftT, bias, fvn);
  k_ppm  <<<dim3(kHW / 32, kN), 256, 0, stream>>>(fnT, fvn, out);
}
